// DGCNN_Block_87436944212103
// MI455X (gfx1250) — matrix operand order repaired and checked by reading, not compiled, not run on hardware
//
#include <hip/hip_runtime.h>
#include <hip/hip_bf16.h>

#define Bz 4
#define Cz 128
#define Nz 4096
#define Kz 9
#define Gz 4
#define FP 132   // padded featL pitch (bank-conflict avoidance)

typedef __attribute__((ext_vector_type(2)))  float  v2f;
typedef __attribute__((ext_vector_type(8)))  float  v8f;
typedef __attribute__((ext_vector_type(16))) __bf16 v16bf;

// ---------------- branchless top-k insert (registers only) ----------------
__device__ __forceinline__ void topk_insert(float (&tv)[Kz], int (&ti)[Kz], float v, int id) {
    if (v <= tv[Kz - 1]) return;
    int rank = 0;
#pragma unroll
    for (int q = 0; q < Kz; ++q) rank += (tv[q] >= v) ? 1 : 0;
#pragma unroll
    for (int p = Kz - 1; p > 0; --p) {
        bool mv = (p > rank);
        tv[p] = mv ? tv[p - 1] : tv[p];
        ti[p] = mv ? ti[p - 1] : ti[p];
    }
#pragma unroll
    for (int q = 0; q < Kz; ++q)
        if (q == rank) { tv[q] = v; ti[q] = id; }
}

// ---------------- prep kernels ----------------
__global__ void prep_transpose(const float* __restrict__ x, float* __restrict__ xt) {
    size_t e = (size_t)blockIdx.x * 256 + threadIdx.x;            // over B*C*N
    int n = (int)(e & (Nz - 1));
    int c = (int)((e >> 12) & (Cz - 1));
    int b = (int)(e >> 19);
    xt[((size_t)b * Nz + n) * Cz + c] = x[e];
}

__global__ void prep_xx(const float* __restrict__ x, float* __restrict__ xx) {
    int t = blockIdx.x * 256 + threadIdx.x;                       // over B*N
    int b = t >> 12, n = t & (Nz - 1);
    const float* xb = x + (size_t)b * Cz * Nz + n;
    float s = 0.f;
    for (int c = 0; c < Cz; ++c) { float v = xb[(size_t)c * Nz]; s += v * v; }
    xx[t] = s;
}

// pack w_lin[128][256] (f32) into bf16 WMMA A-operand tiles:
// element j of lane's v16bf:  K = hi*16 + j  (contiguous k order; B operand uses the same)
__global__ void prep_wpack(const float* __restrict__ w_lin, __bf16* __restrict__ wpack) {
    int t = blockIdx.x * 256 + threadIdx.x;                       // 32768
    int j = t & 15, lane = (t >> 4) & 31, tile = t >> 9;
    int kt = tile & 7, mt = tile >> 3;
    int hi = lane >> 4, m = mt * 16 + (lane & 15);
    int k = kt * 32 + hi * 16 + j;
    wpack[t] = (__bf16)w_lin[m * 256 + k];
}

// pack conv_w[128][128] into f32 WMMA 16x16x4 A-operand pairs
__global__ void prep_cwpack(const float* __restrict__ conv_w, float2* __restrict__ cwpack) {
    int t = blockIdx.x * 256 + threadIdx.x;                       // 8192
    int lane = t & 31, kt = (t >> 5) & 31, mt = t >> 10;
    int hi = lane >> 4, m = mt * 16 + (lane & 15);
    int c = kt * 4 + hi * 2;
    cwpack[t] = make_float2(conv_w[m * 128 + c], conv_w[m * 128 + c + 1]);
}

// ---------------- fused KNN: xT.x GEMM (f32 WMMA) + streaming top-9 ----------------
__global__ __launch_bounds__(128) void knn_kernel(const float* __restrict__ x,
                                                  const float* __restrict__ xx,
                                                  int* __restrict__ idxOut) {
    __shared__ float tileS[4][16][17];
    __shared__ float mVal[4 * 2 * 16 * Kz];
    __shared__ int   mIdx[4 * 2 * 16 * Kz];
    const int b   = blockIdx.x >> 8;            // N/16 == 256 row-tiles
    const int m0  = (blockIdx.x & 255) << 4;
    const int wv  = threadIdx.x >> 5;
    const int lane = threadIdx.x & 31;
    const int ln  = lane & 15;
    const int hi  = lane >> 4;
    const float* xb  = x  + (size_t)b * Cz * Nz;
    const float* xxb = xx + b * Nz;

    // cache A operand (16 rows x 128 channels) in VGPRs: 32 k-steps of float2
    v2f aReg[32];
#pragma unroll
    for (int kt = 0; kt < 32; ++kt) {
        int c = kt * 4 + hi * 2;
        aReg[kt].x = xb[(size_t)c * Nz + m0 + ln];
        aReg[kt].y = xb[(size_t)(c + 1) * Nz + m0 + ln];
    }
    float xxr[8];
#pragma unroll
    for (int r = 0; r < 8; ++r) xxr[r] = xxb[m0 + r + hi * 8];

    float tv[Kz]; int ti[Kz];
#pragma unroll
    for (int q = 0; q < Kz; ++q) { tv[q] = -3.4e38f; ti[q] = 0; }

    for (int ct = wv; ct < Nz / 16; ct += 4) {
        int n0 = ct << 4;
        float xxc = xxb[n0 + ln];
        v8f acc = {};
#pragma unroll
        for (int kt = 0; kt < 32; ++kt) {
            int c = kt * 4 + hi * 2;
            v2f bv;
            bv.x = xb[(size_t)c * Nz + n0 + ln];
            bv.y = xb[(size_t)(c + 1) * Nz + n0 + ln];
            acc = __builtin_amdgcn_wmma_f32_16x16x4_f32(false, aReg[kt], false, bv,
                                                        (short)0, acc, false, false);
        }
        __builtin_prefetch(xb + n0 + 64 + ln, 0, 1);   // next column tile for this wave
        // pd = 2*inner - xx[m] - xx[n]; transpose through LDS so lane==row
#pragma unroll
        for (int r = 0; r < 8; ++r)
            tileS[wv][r + hi * 8][ln] = 2.f * acc[r] - xxr[r] - xxc;
        asm volatile("s_wait_dscnt 0" ::: "memory");
#pragma unroll
        for (int j = 0; j < 8; ++j) {
            int col = hi * 8 + j;
            topk_insert(tv, ti, tileS[wv][ln][col], n0 + col);
        }
        asm volatile("s_wait_dscnt 0" ::: "memory");
    }
    // merge 8 partial lists (4 waves x 2 half-scans) per row
    {
        int base = ((wv * 2 + hi) * 16 + ln) * Kz;
#pragma unroll
        for (int q = 0; q < Kz; ++q) { mVal[base + q] = tv[q]; mIdx[base + q] = ti[q]; }
    }
    __syncthreads();
    if (threadIdx.x < 16) {
        int row = threadIdx.x;
        float fv[Kz]; int fi[Kz];
#pragma unroll
        for (int q = 0; q < Kz; ++q) { fv[q] = -3.4e38f; fi[q] = 0; }
        for (int L = 0; L < 8; ++L) {
            int base = (L * 16 + row) * Kz;
            for (int q = 0; q < Kz; ++q) topk_insert(fv, fi, mVal[base + q], mIdx[base + q]);
        }
        int ob = (b * Nz + m0 + row) * Kz;
#pragma unroll
        for (int q = 0; q < Kz; ++q) idxOut[ob + q] = fi[q];
    }
}

// ---------------- edge linear (bf16 WMMA) + grouped attention ----------------
__global__ __launch_bounds__(128) void edge_attn_kernel(
        const float* __restrict__ x, const float* __restrict__ xt,
        const int* __restrict__ idxG, const __bf16* __restrict__ wpack,
        const float* __restrict__ b_lin, const float* __restrict__ w_aw,
        const float* __restrict__ b_aw, float* __restrict__ yBuf) {
    __shared__ __align__(16) __bf16 edgeB[4][16 * 256];   // B operand [col][K], bf16
    __shared__ float featL[4][16 * FP];                   // feat [col][chan]
    __shared__ float ctrF[4][Cz];
    __shared__ float awS[4][40];
    __shared__ float scoresL[4][81];
    __shared__ float wS[4][12];

    const int wv = threadIdx.x >> 5;
    const int lane = threadIdx.x & 31;
    const int ln = lane & 15;
    const int hi = lane >> 4;
    const int p = blockIdx.x * 4 + wv;       // one point per wave
    const int b = p >> 12;
    const int n = p & (Nz - 1);

    const float* ctrRow = xt + (size_t)(b * Nz + n) * Cz;
    float ctrReg[4];
#pragma unroll
    for (int t = 0; t < 4; ++t) { ctrReg[t] = ctrRow[t * 32 + lane]; ctrF[wv][t * 32 + lane] = ctrReg[t]; }

    // build edge = [ctr ; ctr - nbr] per neighbor column, bf16, pad cols 9..15 with 0
    for (int k = 0; k < Kz; ++k) {
        int nb = idxG[(b * Nz + n) * Kz + k];
        const float* nbRow = xt + (size_t)(b * Nz + nb) * Cz;
#pragma unroll
        for (int t = 0; t < 4; ++t) {
            int c = t * 32 + lane;
            float nv = nbRow[c];
            edgeB[wv][k * 256 + c]       = (__bf16)ctrReg[t];
            edgeB[wv][k * 256 + 128 + c] = (__bf16)(ctrReg[t] - nv);
        }
    }
#pragma unroll
    for (int k = Kz; k < 16; ++k)
#pragma unroll
        for (int t = 0; t < 8; ++t)
            edgeB[wv][k * 256 + t * 32 + lane] = (__bf16)0.f;
    asm volatile("s_wait_dscnt 0" ::: "memory");

    // feat[col][m] = sum_e w_lin[m][e] * edge[col][e]  (bf16 WMMA, K=256 -> 8 steps)
#pragma unroll 1
    for (int mt = 0; mt < 8; ++mt) {
        v8f acc = {};
#pragma unroll
        for (int kt = 0; kt < 8; ++kt) {
            v16bf av = *(const v16bf*)(wpack + (size_t)((mt * 8 + kt) * 32 + lane) * 16);
            v16bf bv = *(const v16bf*)(&edgeB[wv][ln * 256 + kt * 32 + hi * 16]);
            acc = __builtin_amdgcn_wmma_f32_16x16x32_bf16(false, av, false, bv,
                                                          (short)0, acc, false, false);
        }
#pragma unroll
        for (int r = 0; r < 8; ++r) {
            int m = mt * 16 + r + hi * 8;
            featL[wv][ln * FP + m] = acc[r] + b_lin[m];
        }
    }
    asm volatile("s_wait_dscnt 0" ::: "memory");

    // ---- neighbor-attention weights aw = softmax_k(w_aw @ ctr + b_aw) ----
    {
        float acc = b_aw[lane];
        const float* wr = w_aw + lane * Cz;
#pragma unroll 4
        for (int c = 0; c < Cz; ++c) acc += wr[c] * ctrF[wv][c];
        awS[wv][lane] = acc;
        if (lane < 4) {
            int d = 32 + lane;
            float a2 = b_aw[d];
            const float* wr2 = w_aw + d * Cz;
#pragma unroll 4
            for (int c = 0; c < Cz; ++c) a2 += wr2[c] * ctrF[wv][c];
            awS[wv][d] = a2;
        }
    }
    asm volatile("s_wait_dscnt 0" ::: "memory");
    if (lane < 4) {
        int g = lane;
        float mx = -3.4e38f; float e[Kz];
#pragma unroll
        for (int k2 = 0; k2 < Kz; ++k2) mx = fmaxf(mx, awS[wv][g * Kz + k2]);
        float sm = 0.f;
#pragma unroll
        for (int k2 = 0; k2 < Kz; ++k2) { e[k2] = __expf(awS[wv][g * Kz + k2] - mx); sm += e[k2]; }
        float inv = 1.f / sm;
#pragma unroll
        for (int k2 = 0; k2 < Kz; ++k2) awS[wv][g * Kz + k2] = e[k2] * inv;
    }
    asm volatile("s_wait_dscnt 0" ::: "memory");

    const float invs = 0.17677669529663687f;  // 1/sqrt(32)
#pragma unroll 1
    for (int g = 0; g < Gz; ++g) {
        int gb = g * 32;
        // 9x9 scores
#pragma unroll 1
        for (int t = 0; t < 3; ++t) {
            int pid = t * 32 + lane;
            if (pid < 81) {
                int k2 = pid / 9, j = pid % 9;
                const float* fk = &featL[wv][k2 * FP + gb];
                const float* fj = &featL[wv][j * FP + gb];
                float s = 0.f;
#pragma unroll
                for (int cc = 0; cc < 32; ++cc) s += fk[cc] * fj[cc];
                scoresL[wv][pid] = s * invs;
            }
        }
        asm volatile("s_wait_dscnt 0" ::: "memory");
        if (lane < Kz) {            // softmax rows
            int k2 = lane;
            float mx = -3.4e38f; float e[Kz];
#pragma unroll
            for (int j = 0; j < Kz; ++j) mx = fmaxf(mx, scoresL[wv][k2 * 9 + j]);
            float sm = 0.f;
#pragma unroll
            for (int j = 0; j < Kz; ++j) { e[j] = __expf(scoresL[wv][k2 * 9 + j] - mx); sm += e[j]; }
            float inv = 1.f / sm;
#pragma unroll
            for (int j = 0; j < Kz; ++j) scoresL[wv][k2 * 9 + j] = e[j] * inv;
        }
        asm volatile("s_wait_dscnt 0" ::: "memory");
        if (lane < Kz) {            // fold aw into attention: w_j = sum_k aw_k s_kj
            int j = lane;
            float w = 0.f;
#pragma unroll
            for (int k2 = 0; k2 < Kz; ++k2) w += awS[wv][g * Kz + k2] * scoresL[wv][k2 * 9 + j];
            wS[wv][j] = w;
        }
        asm volatile("s_wait_dscnt 0" ::: "memory");
        {
            float lf = 0.f;
#pragma unroll
            for (int j = 0; j < Kz; ++j) {
                float fv = featL[wv][j * FP + gb + lane];
                float ev = fv > 0.f ? fv : (__expf(fv) - 1.f);
                lf += wS[wv][j] * ev;
            }
            int c = gb + lane;
            size_t oi = (size_t)b * Cz * Nz + (size_t)c * Nz + n;
            yBuf[oi] = lf + x[oi];          // residual
        }
        asm volatile("s_wait_dscnt 0" ::: "memory");
    }
}

// ---------------- GroupNorm stats -> per (b,c) scale/shift ----------------
__global__ __launch_bounds__(256) void gn_stats_kernel(const float* __restrict__ yBuf,
                                                       const float* __restrict__ gn_g,
                                                       const float* __restrict__ gn_b,
                                                       float* __restrict__ sc,
                                                       float* __restrict__ sh) {
    __shared__ float rs[256], rq[256];
    int b = blockIdx.x >> 2, g = blockIdx.x & 3;
    const float* base = yBuf + (size_t)b * Cz * Nz + (size_t)g * 32 * Nz;
    float s = 0.f, q = 0.f;
    for (int t = threadIdx.x; t < 32 * Nz; t += 256) { float v = base[t]; s += v; q += v * v; }
    rs[threadIdx.x] = s; rq[threadIdx.x] = q;
    __syncthreads();
    for (int o = 128; o > 0; o >>= 1) {
        if (threadIdx.x < o) { rs[threadIdx.x] += rs[threadIdx.x + o]; rq[threadIdx.x] += rq[threadIdx.x + o]; }
        __syncthreads();
    }
    if (threadIdx.x < 32) {
        float mean = rs[0] * (1.f / (32.f * Nz));
        float var  = rq[0] * (1.f / (32.f * Nz)) - mean * mean;
        float rstd = rsqrtf(var + 1e-5f);
        int c = g * 32 + threadIdx.x;
        float scv = rstd * gn_g[c];
        sc[b * Cz + c] = scv;
        sh[b * Cz + c] = gn_b[c] - mean * scv;
    }
}

// ---------------- 1x1 conv (f32 WMMA, GN folded) + BN partial sums ----------------
__global__ __launch_bounds__(128) void conv_kernel(const float* __restrict__ yBuf,
                                                   const float2* __restrict__ cwpack,
                                                   const float* __restrict__ conv_b,
                                                   const float* __restrict__ sc,
                                                   const float* __restrict__ sh,
                                                   float* __restrict__ zBuf,
                                                   float* __restrict__ partials) {
    __shared__ float scL[Cz], shL[Cz], wgSum[Cz], wgSq[Cz];
    int q0 = blockIdx.x << 4;
    int b  = q0 >> 12;
    int n0 = q0 & (Nz - 1);
    if (threadIdx.x < Cz) { scL[threadIdx.x] = sc[b * Cz + threadIdx.x]; shL[threadIdx.x] = sh[b * Cz + threadIdx.x]; }
    __syncthreads();
    const int wv = threadIdx.x >> 5, lane = threadIdx.x & 31, ln = lane & 15, hi = lane >> 4;
    const float* yb = yBuf + (size_t)b * Cz * Nz;
    float* zb = zBuf + (size_t)b * Cz * Nz;
#pragma unroll 1
    for (int s2 = 0; s2 < 2; ++s2) {
        int mt = wv * 2 + s2;
        v8f acc = {};
#pragma unroll
        for (int kt = 0; kt < 32; ++kt) {
            float2 aw2 = cwpack[(mt * 32 + kt) * 32 + lane];
            v2f av; av.x = aw2.x; av.y = aw2.y;
            int c = kt * 4 + hi * 2;
            v2f bv;
            bv.x = yb[(size_t)c * Nz + n0 + ln] * scL[c] + shL[c];
            bv.y = yb[(size_t)(c + 1) * Nz + n0 + ln] * scL[c + 1] + shL[c + 1];
            acc = __builtin_amdgcn_wmma_f32_16x16x4_f32(false, av, false, bv,
                                                        (short)0, acc, false, false);
        }
#pragma unroll
        for (int r = 0; r < 8; ++r) {
            int m = mt * 16 + r + hi * 8;
            float zv = acc[r] + conv_b[m];
            zb[(size_t)m * Nz + n0 + ln] = zv;
            float zs = zv, zq = zv * zv;
#pragma unroll
            for (int o = 1; o < 16; o <<= 1) { zs += __shfl_xor(zs, o, 16); zq += __shfl_xor(zq, o, 16); }
            if (ln == 0) { wgSum[m] = zs; wgSq[m] = zq; }
        }
    }
    __syncthreads();
    if (threadIdx.x < Cz) {
        partials[(size_t)blockIdx.x * 256 + threadIdx.x]      = wgSum[threadIdx.x];
        partials[(size_t)blockIdx.x * 256 + Cz + threadIdx.x] = wgSq[threadIdx.x];
    }
}

// ---------------- BN reduce (deterministic) ----------------
__global__ __launch_bounds__(256) void bn_reduce_kernel(const float* __restrict__ partials,
                                                        const float* __restrict__ bn_g,
                                                        const float* __restrict__ bn_b,
                                                        float* __restrict__ bnsc,
                                                        float* __restrict__ bnsh) {
    __shared__ float sums[256];
    int tid = threadIdx.x;
    float s = 0.f;
    for (int w = 0; w < (Bz * Nz) / 16; ++w) s += partials[(size_t)w * 256 + tid];
    sums[tid] = s;
    __syncthreads();
    if (tid < Cz) {
        float mean = sums[tid] * (1.f / (Bz * Nz));
        float var  = sums[Cz + tid] * (1.f / (Bz * Nz)) - mean * mean;
        float scv  = bn_g[tid] * rsqrtf(var + 1e-5f);
        bnsc[tid] = scv;
        bnsh[tid] = bn_b[tid] - mean * scv;
    }
}

// ---------------- BN apply + ReLU ----------------
__global__ __launch_bounds__(256) void bn_apply_kernel(const float* __restrict__ zBuf,
                                                       const float* __restrict__ bnsc,
                                                       const float* __restrict__ bnsh,
                                                       float* __restrict__ out) {
    size_t i = ((size_t)blockIdx.x * 256 + threadIdx.x) * 4;
    int c = (int)((i >> 12) & (Cz - 1));
    float4 z = *(const float4*)(zBuf + i);
    float a = bnsc[c], d = bnsh[c];
    float4 r;
    r.x = fmaxf(z.x * a + d, 0.f);
    r.y = fmaxf(z.y * a + d, 0.f);
    r.z = fmaxf(z.z * a + d, 0.f);
    r.w = fmaxf(z.w * a + d, 0.f);
    *(float4*)(out + i) = r;
}

// ---------------- host launcher ----------------
extern "C" void kernel_launch(void* const* d_in, const int* in_sizes, int n_in,
                              void* d_out, int out_size, void* d_ws, size_t ws_size,
                              hipStream_t stream) {
    (void)in_sizes; (void)n_in; (void)out_size; (void)ws_size;
    const float* features = (const float*)d_in[0];
    const float* w_lin  = (const float*)d_in[1];
    const float* b_lin  = (const float*)d_in[2];
    const float* w_aw   = (const float*)d_in[3];
    const float* b_aw   = (const float*)d_in[4];
    const float* gn_g   = (const float*)d_in[5];
    const float* gn_b   = (const float*)d_in[6];
    const float* conv_w = (const float*)d_in[7];
    const float* conv_b = (const float*)d_in[8];
    const float* bn_g   = (const float*)d_in[9];
    const float* bn_b   = (const float*)d_in[10];

    char* ws = (char*)d_ws;
    size_t o = 0;
    auto take = [&](size_t bytes) { char* r = ws + o; o += (bytes + 255) & ~(size_t)255; return r; };
    float*  xt    = (float*) take((size_t)Bz * Nz * Cz * 4);
    float*  xx    = (float*) take((size_t)Bz * Nz * 4);
    __bf16* wpack = (__bf16*)take((size_t)64 * 32 * 16 * 2);
    float2* cwpk  = (float2*)take((size_t)8 * 32 * 32 * 8);
    int*    idx   = (int*)   take((size_t)Bz * Nz * Kz * 4);
    float*  yBuf  = (float*) take((size_t)Bz * Cz * Nz * 4);
    float*  zBuf  = (float*) take((size_t)Bz * Cz * Nz * 4);
    float*  sc    = (float*) take((size_t)Bz * Cz * 4);
    float*  sh    = (float*) take((size_t)Bz * Cz * 4);
    float*  parts = (float*) take((size_t)(Bz * Nz / 16) * 256 * 4);
    float*  bnsc  = (float*) take((size_t)Cz * 4);
    float*  bnsh  = (float*) take((size_t)Cz * 4);

    prep_transpose<<<(Bz * Cz * Nz) / 256, 256, 0, stream>>>(features, xt);
    prep_xx<<<(Bz * Nz) / 256, 256, 0, stream>>>(features, xx);
    prep_wpack<<<128, 256, 0, stream>>>(w_lin, wpack);
    prep_cwpack<<<32, 256, 0, stream>>>(conv_w, cwpk);
    knn_kernel<<<Bz * (Nz / 16), 128, 0, stream>>>(features, xx, idx);
    edge_attn_kernel<<<(Bz * Nz) / 4, 128, 0, stream>>>(features, xt, idx, wpack,
                                                        b_lin, w_aw, b_aw, yBuf);
    gn_stats_kernel<<<Bz * Gz, 256, 0, stream>>>(yBuf, gn_g, gn_b, sc, sh);
    conv_kernel<<<(Bz * Nz) / 16, 128, 0, stream>>>(yBuf, cwpk, conv_b, sc, sh, zBuf, parts);
    bn_reduce_kernel<<<1, 256, 0, stream>>>(parts, bn_g, bn_b, bnsc, bnsh);
    bn_apply_kernel<<<(Bz * Cz * Nz) / 1024, 256, 0, stream>>>(zBuf, bnsc, bnsh, (float*)d_out);
}
